// GCNModel_42958262895197
// MI455X (gfx1250) — matrix operand order repaired and checked by reading, not compiled, not run on hardware
//
#include <hip/hip_runtime.h>
#include <hip/hip_bf16.h>

// ---------------- WMMA vector types (gfx1250, wave32) ----------------
typedef _Float16 v8h  __attribute__((ext_vector_type(8)));
typedef _Float16 v16h __attribute__((ext_vector_type(16)));
typedef float    v8f  __attribute__((ext_vector_type(8)));

#define N_NODES 100000
#define N_EDGES 800000
#define N_GRAPH 4096
#define F_IN    143
#define HDIM    128
#define KPAD1   160            // 143 padded up to multiple of 32
#define BN_INV  0.99999500003749963f   // rsqrt(1 + 1e-5)

// ---------------- tiny elementwise helpers ----------------
__global__ void k_fill_f32(float* __restrict__ p, float v, int n) {
    int i = blockIdx.x * 256 + threadIdx.x;
    if (i < n) p[i] = v;
}

__global__ void k_count_deg(const int* __restrict__ dst, float* __restrict__ deg, int e) {
    int i = blockIdx.x * 256 + threadIdx.x;
    if (i < e) unsafeAtomicAdd(deg + dst[i], 1.0f);
}

__global__ void k_rsqrt_inplace(float* __restrict__ p, int n) {
    int i = blockIdx.x * 256 + threadIdx.x;
    if (i < n) p[i] = rsqrtf(p[i]);
}

// edge_norm[e] = dis[src[e]] * dis[dst[e]]
__global__ void k_edge_norm(const int* __restrict__ src, const int* __restrict__ dst,
                            const float* __restrict__ dis, float* __restrict__ en, int e) {
    int i = blockIdx.x * 256 + threadIdx.x;
    if (i < e) en[i] = dis[src[i]] * dis[dst[i]];
}

// x [N, F_IN] f32 -> hA [N, KPAD1] f16, zero-padded K
__global__ void k_cvt_x(const float* __restrict__ x, _Float16* __restrict__ hA, int total) {
    int i = blockIdx.x * 256 + threadIdx.x;
    if (i >= total) return;
    int row = i / KPAD1, k = i - row * KPAD1;
    hA[i] = (k < F_IN) ? (_Float16)x[row * F_IN + k] : (_Float16)0.0f;
}

// W [K, H] row-major f32 -> Wt [H, Kpad] f16 (column-major: K contiguous per out channel)
__global__ void k_cvt_w(const float* __restrict__ W, _Float16* __restrict__ Wt, int K, int Kpad) {
    int i = blockIdx.x * 256 + threadIdx.x;
    if (i >= HDIM * Kpad) return;
    int n = i / Kpad, k = i - n * Kpad;
    Wt[i] = (k < K) ? (_Float16)W[k * HDIM + n] : (_Float16)0.0f;
}

// ---------------- WMMA GEMM: out[N,H] = A[N,Kpad] @ Wt^T ----------------
// One wave computes a 16x32 output tile (two accumulators share the A fragment):
//   A frag (contiguous k order, not the ISA table order): lane l -> m = l&15, kb = (l>>4)*16,
//     halfs 0..15 = K[kb..kb+15]   (two 16B loads)
//   B frag (same contiguous k order as A, rows striped across lanes): lane l -> n = l&15,
//     K = (l>>4)*16 + [0..15], contiguous in column-major Wt    (two 16B loads each)
// 8 waves/block; block covers 32 rows x 128 cols. KPAD compile-time => full unroll.
template <int KPAD>
__global__ __launch_bounds__(256) void k_gemm_wmma(
    const _Float16* __restrict__ A,   // [Mtiles*16, KPAD]
    const _Float16* __restrict__ Bt,  // [HDIM, KPAD]
    float* __restrict__ out)          // [Mtiles*16, HDIM]
{
    const int wave   = threadIdx.x >> 5;
    const int lane   = threadIdx.x & 31;
    const int tile   = blockIdx.x * 8 + wave;   // N/16 * 4 tiles total
    const int tm     = tile >> 2;               // row tile (16 rows)
    const int tn2    = tile & 3;                // 32-col group (H/32 = 4)
    const int half16 = lane >> 4;               // 0 or 1
    const int lane15 = lane & 15;

    const _Float16* ap  = A  + (size_t)(tm * 16 + lane15) * KPAD + half16 * 16;
    const _Float16* bp0 = Bt + (size_t)(tn2 * 32 + lane15) * KPAD + half16 * 16;
    const _Float16* bp1 = bp0 + (size_t)16 * KPAD;   // next 16 output channels

    v8f acc0 = {};
    v8f acc1 = {};
    #pragma unroll
    for (int k0 = 0; k0 < KPAD; k0 += 32) {
        v8h a0 = *(const v8h*)(ap + k0);
        v8h a1 = *(const v8h*)(ap + k0 + 8);
        v16h av = __builtin_shufflevector(a0, a1, 0,1,2,3,4,5,6,7,8,9,10,11,12,13,14,15);

        v8h b00 = *(const v8h*)(bp0 + k0);
        v8h b01 = *(const v8h*)(bp0 + k0 + 8);
        v16h bv0 = __builtin_shufflevector(b00, b01, 0,1,2,3,4,5,6,7,8,9,10,11,12,13,14,15);
        acc0 = __builtin_amdgcn_wmma_f32_16x16x32_f16(false, av, false, bv0,
                                                      (short)0, acc0, false, false);

        v8h b10 = *(const v8h*)(bp1 + k0);
        v8h b11 = *(const v8h*)(bp1 + k0 + 8);
        v16h bv1 = __builtin_shufflevector(b10, b11, 0,1,2,3,4,5,6,7,8,9,10,11,12,13,14,15);
        acc1 = __builtin_amdgcn_wmma_f32_16x16x32_f16(false, av, false, bv1,
                                                      (short)0, acc1, false, false);
    }

    // C/D layout: VGPR r -> row tm*16 + r + half16*8, col base + lane15
    float* op = out + (size_t)(tm * 16 + half16 * 8) * HDIM + tn2 * 32 + lane15;
    #pragma unroll
    for (int r = 0; r < 8; ++r) {
        op[(size_t)r * HDIM]      = acc0[r];
        op[(size_t)r * HDIM + 16] = acc1[r];
    }
}

// ---------------- edge scatter: one wave32 per edge, float4 per lane ----------------
// agg[dst] += lin[src] * edge_norm[e];  128 channels = 32 lanes x float4
__global__ __launch_bounds__(256) void k_scatter(
    const float* __restrict__ lin, const float* __restrict__ en,
    const int* __restrict__ src, const int* __restrict__ dst,
    float* __restrict__ agg, int total_waveslots)
{
    int gid = blockIdx.x * 256 + threadIdx.x;
    if (gid >= total_waveslots) return;
    int e    = gid >> 5;
    int lane = gid & 31;
    int s = src[e], d = dst[e];
    float w = en[e];
    const float4 v = *(const float4*)(lin + (size_t)s * HDIM + lane * 4);
    float* ag = agg + (size_t)d * HDIM + lane * 4;
    unsafeAtomicAdd(ag + 0, v.x * w);
    unsafeAtomicAdd(ag + 1, v.y * w);
    unsafeAtomicAdd(ag + 2, v.z * w);
    unsafeAtomicAdd(ag + 3, v.w * w);
}

// ---------------- bias + self-loop + BN(eval) + ReLU; emit f32 + f16 ----------------
__global__ __launch_bounds__(256) void k_finalize(
    const float* __restrict__ agg, const float* __restrict__ lin,
    const float* __restrict__ dis, const float* __restrict__ b,
    const float* __restrict__ gamma, const float* __restrict__ beta,
    float* __restrict__ hf, _Float16* __restrict__ hA, int total)
{
    int gid = blockIdx.x * 256 + threadIdx.x;
    if (gid >= total) return;
    int i = gid >> 7;
    int c = gid & 127;
    float sn = dis[i]; sn *= sn;
    float v = agg[gid] + lin[gid] * sn + b[c];
    v = gamma[c] * (v * BN_INV) + beta[c];
    v = fmaxf(v, 0.0f);
    hf[gid] = v;
    hA[gid] = (_Float16)v;     // stride-128 f16 input for next layer's GEMM
}

// ---------------- per-graph mean pool (sums + counts via atomics) ----------------
__global__ __launch_bounds__(256) void k_pool_sum(
    const float* __restrict__ hf, const int* __restrict__ batch,
    float* __restrict__ pool, float* __restrict__ cnt, int total)
{
    int gid = blockIdx.x * 256 + threadIdx.x;
    if (gid >= total) return;
    int i = gid >> 7;
    int c = gid & 127;
    int g = batch[i];
    unsafeAtomicAdd(pool + (size_t)g * HDIM + c, hf[gid]);
    if (c == 0) unsafeAtomicAdd(cnt + g, 1.0f);
}

// ---------------- MLP head: out[g] = relu(mean@wc1+bc1)@wc2 + bc2 ----------------
__global__ __launch_bounds__(64) void k_mlp(
    const float* __restrict__ pool, const float* __restrict__ cnt,
    const float* __restrict__ wc1, const float* __restrict__ bc1,
    const float* __restrict__ wc2, const float* __restrict__ bc2,
    float* __restrict__ out)
{
    __shared__ float sm[HDIM];
    __shared__ float sr[64];
    int g = blockIdx.x, t = threadIdx.x;
    float c = fmaxf(cnt[g], 1.0f);
    sm[t]      = pool[(size_t)g * HDIM + t] / c;
    sm[t + 64] = pool[(size_t)g * HDIM + t + 64] / c;
    __syncthreads();
    float acc = bc1[t];
    #pragma unroll 8
    for (int k = 0; k < HDIM; ++k) acc += sm[k] * wc1[k * 64 + t];
    acc = fmaxf(acc, 0.0f);
    sr[t] = acc * wc2[t];
    __syncthreads();
    for (int s = 32; s > 0; s >>= 1) {
        if (t < s) sr[t] += sr[t + s];
        __syncthreads();
    }
    if (t == 0) out[g] = sr[0] + bc2[0];
}

// ---------------- launcher ----------------
extern "C" void kernel_launch(void* const* d_in, const int* in_sizes, int n_in,
                              void* d_out, int out_size, void* d_ws, size_t ws_size,
                              hipStream_t stream) {
    const float* x      = (const float*)d_in[0];
    const int*   ei     = (const int*)d_in[1];     // [2,E]: src = ei, dst = ei+E
    const int*   batch  = (const int*)d_in[2];
    const float* w0     = (const float*)d_in[3];
    const float* b0     = (const float*)d_in[4];
    const float* wsL    = (const float*)d_in[5];   // [3,H,H]
    const float* bsL    = (const float*)d_in[6];   // [3,H]
    const float* gammas = (const float*)d_in[7];   // [4,H]
    const float* betas  = (const float*)d_in[8];   // [4,H]
    const float* wc1    = (const float*)d_in[9];
    const float* bc1    = (const float*)d_in[10];
    const float* wc2    = (const float*)d_in[11];
    const float* bc2    = (const float*)d_in[12];
    float* out = (float*)d_out;

    const int* e_src = ei;
    const int* e_dst = ei + N_EDGES;

    // ---- workspace carve-up ----
    char* wsb = (char*)d_ws;
    size_t off = 0;
    auto carve = [&](size_t bytes) { size_t r = off; off += (bytes + 255) & ~(size_t)255; return r; };
    float*    dis  = (float*)   (wsb + carve((size_t)N_NODES * 4));
    float*    en   = (float*)   (wsb + carve((size_t)N_EDGES * 4));
    _Float16* hA   = (_Float16*)(wsb + carve((size_t)N_NODES * KPAD1 * 2));
    _Float16* Wt   = (_Float16*)(wsb + carve((size_t)HDIM * KPAD1 * 2));
    float*    lin  = (float*)   (wsb + carve((size_t)N_NODES * HDIM * 4));
    float*    agg  = (float*)   (wsb + carve((size_t)N_NODES * HDIM * 4));
    float*    hf   = (float*)   (wsb + carve((size_t)N_NODES * HDIM * 4));
    float*    pool = (float*)   (wsb + carve((size_t)N_GRAPH * HDIM * 4));
    float*    cnt  = (float*)   (wsb + carve((size_t)N_GRAPH * 4));

    const int NH = N_NODES * HDIM;          // 12.8M
    const int EW = N_EDGES * 32;            // wave-per-edge slots: 25.6M
    const int GEMM_BLOCKS = (N_NODES / 16) * 4 / 8;   // 3125
    auto blocks = [](int n) { return (n + 255) / 256; };

    // degrees -> dis = deg^{-1/2} (with self-loop), then per-edge norms
    k_fill_f32<<<blocks(N_NODES), 256, 0, stream>>>(dis, 1.0f, N_NODES);
    k_count_deg<<<blocks(N_EDGES), 256, 0, stream>>>(e_dst, dis, N_EDGES);
    k_rsqrt_inplace<<<blocks(N_NODES), 256, 0, stream>>>(dis, N_NODES);
    k_edge_norm<<<blocks(N_EDGES), 256, 0, stream>>>(e_src, e_dst, dis, en, N_EDGES);

    // ---- layer 0 (K=143 -> Kpad 160) ----
    k_cvt_x<<<blocks(N_NODES * KPAD1), 256, 0, stream>>>(x, hA, N_NODES * KPAD1);
    k_cvt_w<<<blocks(HDIM * KPAD1), 256, 0, stream>>>(w0, Wt, F_IN, KPAD1);
    k_gemm_wmma<KPAD1><<<GEMM_BLOCKS, 256, 0, stream>>>(hA, Wt, lin);
    k_fill_f32<<<blocks(NH), 256, 0, stream>>>(agg, 0.0f, NH);
    k_scatter<<<blocks(EW), 256, 0, stream>>>(lin, en, e_src, e_dst, agg, EW);
    k_finalize<<<blocks(NH), 256, 0, stream>>>(agg, lin, dis, b0, gammas, betas, hf, hA, NH);

    // ---- layers 1..3 (K=128) ----
    for (int l = 1; l < 4; ++l) {
        k_cvt_w<<<blocks(HDIM * HDIM), 256, 0, stream>>>(wsL + (size_t)(l - 1) * HDIM * HDIM, Wt, HDIM, HDIM);
        k_gemm_wmma<HDIM><<<GEMM_BLOCKS, 256, 0, stream>>>(hA, Wt, lin);
        k_fill_f32<<<blocks(NH), 256, 0, stream>>>(agg, 0.0f, NH);
        k_scatter<<<blocks(EW), 256, 0, stream>>>(lin, en, e_src, e_dst, agg, EW);
        k_finalize<<<blocks(NH), 256, 0, stream>>>(agg, lin, dis,
            bsL + (size_t)(l - 1) * HDIM, gammas + (size_t)l * HDIM, betas + (size_t)l * HDIM,
            hf, hA, NH);
    }

    // ---- global mean pool + MLP head ----
    k_fill_f32<<<blocks(N_GRAPH * HDIM), 256, 0, stream>>>(pool, 0.0f, N_GRAPH * HDIM);
    k_fill_f32<<<blocks(N_GRAPH), 256, 0, stream>>>(cnt, 0.0f, N_GRAPH);
    k_pool_sum<<<blocks(NH), 256, 0, stream>>>(hf, batch, pool, cnt, NH);
    k_mlp<<<N_GRAPH, 64, 0, stream>>>(pool, cnt, wc1, bc1, wc2, bc2, out);
}
